// SwinUNetBlock_62122406969676
// MI455X (gfx1250) — matrix operand order repaired and checked by reading, not compiled, not run on hardware
//
#include <hip/hip_runtime.h>
#include <hip/hip_bf16.h>

typedef __attribute__((ext_vector_type(16))) _Float16 v16h;
typedef __attribute__((ext_vector_type(8)))  _Float16 v8h;
typedef __attribute__((ext_vector_type(8)))  float    v8f;

constexpr int C_B = 2, C_T = 8, C_H = 60, C_Wd = 60, C_C = 256;
constexpr int C_HEADS = 8, C_HD = 32;
constexpr int C_WH = 5, C_WW = 6, C_SH = 2, C_SW = 3;
constexpr int C_N = 30, C_NW = 120, C_HID = 1024;
constexpr int C_ROWS = C_B * C_NW * C_T * C_N;   // 57600
constexpr int C_BWT  = C_B * C_NW * C_T;          // 1920
constexpr float C_SCALE = 0.17677669529663687f;   // 1/sqrt(32)

// ---------------- WMMA helpers ----------------

__device__ __forceinline__ v8f wmma16(v16h a, v16h b, v8f c) {
  return __builtin_amdgcn_wmma_f32_16x16x32_f16(false, a, false, b, (short)0, c,
                                                false, false);
}

// A fragment (16x32 f16, row-major, row stride ld). Contiguous k order (B operands use the same):
// lane<16: row=lane, K = [0..15]
// lane>=16: row=lane-16, K = [16..31]
__device__ __forceinline__ v16h load_a_frag(const _Float16* base, int ld, int lane) {
  const _Float16* p = base + (size_t)(lane & 15) * ld + ((lane >> 4) << 4);
  v8h lo = *(const v8h*)p;
  v8h hi = *(const v8h*)(p + 8);
  v16h a;
#pragma unroll
  for (int e = 0; e < 8; ++e) { a[e] = lo[e]; a[e + 8] = hi[e]; }
  return a;
}

// B fragment (32x16 f16) from K-contiguous storage: element(k,col) = base[col*ld + k]
// lane<16: col=lane, K=0..15 ; lane>=16: col=lane-16, K=16..31  (contiguous 32B)
__device__ __forceinline__ v16h load_b_kcont(const _Float16* base, int ld, int lane) {
  return *(const v16h*)(base + (size_t)(lane & 15) * ld + ((lane >> 4) << 4));
}

// ---------------- weight pre-pack (fp32 -> f16 fragment-native) ----------------
// packed[((nt16*KT + kt)*32 + lane)*16 + e] = W[kt*32 + (lane>=16?16:0) + e][nt16*16 + (lane&15)]
__global__ void pack_w(const float* __restrict__ src, _Float16* __restrict__ dst,
                       int KT, int Ndim, int total) {
  int t = blockIdx.x * 256 + threadIdx.x;
  if (t >= total) return;
  int e    = t & 15;
  int lane = (t >> 4) & 31;
  int kt   = (t >> 9) % KT;
  int nt16 = t / (KT << 9);
  int col  = nt16 * 16 + (lane & 15);
  int krow = kt * 32 + ((lane >> 4) << 4) + e;
  dst[t] = (_Float16)src[(size_t)krow * Ndim + col];
}

// ---------------- LN1 + cyclic shift + window partition + pos-embed ----------------
__global__ void ln1_partition(const float* __restrict__ x, const float* __restrict__ ce,
                              const float* __restrict__ g1, const float* __restrict__ be1,
                              _Float16* __restrict__ wqk, _Float16* __restrict__ w16) {
  __shared__ float sh0[8], sh1[8];
  const int r = blockIdx.x, tid = threadIdx.x;
  int bwt = r / C_N, n = r - bwt * C_N;
  int b = bwt / (C_NW * C_T), r2 = bwt - b * (C_NW * C_T);
  int nwi = r2 >> 3, t = r2 & 7;
  int whi = nwi / 10, wwi = nwi - whi * 10;
  int i = n / 6, j = n - i * 6;
  int h = whi * C_WH + i + C_SH; if (h >= C_H) h -= C_H;
  int w = wwi * C_WW + j + C_SW; if (w >= C_Wd) w -= C_Wd;
  size_t src = ((((size_t)b * C_T + t) * C_H + h) * C_Wd + w) * C_C;
  float val = x[src + tid];
  float s = val, q = val * val;
#pragma unroll
  for (int o = 16; o > 0; o >>= 1) { s += __shfl_xor(s, o, 32); q += __shfl_xor(q, o, 32); }
  if ((tid & 31) == 0) { sh0[tid >> 5] = s; sh1[tid >> 5] = q; }
  __syncthreads();
  float sum = 0.f, sq = 0.f;
#pragma unroll
  for (int u = 0; u < 8; ++u) { sum += sh0[u]; sq += sh1[u]; }
  float mean = sum * (1.0f / C_C);
  float var  = sq * (1.0f / C_C) - mean * mean;
  float rstd = rsqrtf(var + 1e-5f);
  float xn = (val - mean) * rstd * g1[tid] + be1[tid];
  size_t o2 = (size_t)r * C_C + tid;
  w16[o2] = (_Float16)xn;
  wqk[o2] = (_Float16)(xn + ce[((size_t)b * C_T + t) * C_C + tid]);
}

// ---------------- LN2 ----------------
__global__ void ln2_k(const float* __restrict__ x1, const float* __restrict__ g2,
                      const float* __restrict__ be2, _Float16* __restrict__ xn16) {
  __shared__ float sh0[8], sh1[8];
  const int r = blockIdx.x, tid = threadIdx.x;
  float val = x1[(size_t)r * C_C + tid];
  float s = val, q = val * val;
#pragma unroll
  for (int o = 16; o > 0; o >>= 1) { s += __shfl_xor(s, o, 32); q += __shfl_xor(q, o, 32); }
  if ((tid & 31) == 0) { sh0[tid >> 5] = s; sh1[tid >> 5] = q; }
  __syncthreads();
  float sum = 0.f, sq = 0.f;
#pragma unroll
  for (int u = 0; u < 8; ++u) { sum += sh0[u]; sq += sh1[u]; }
  float mean = sum * (1.0f / C_C);
  float rstd = rsqrtf(sq * (1.0f / C_C) - mean * mean + 1e-5f);
  xn16[(size_t)r * C_C + tid] = (_Float16)((val - mean) * rstd * g2[tid] + be2[tid]);
}

// ---------------- generic WMMA GEMM, one wave = 32x32 tile ----------------
// EPI 0: Q/K scatter -> [bwt][head][tok32][hd32]
// EPI 1: V scatter transposed -> [bwt][head][hd32][tok32]
// EPI 2: O: window-reverse + roll + shortcut -> x1 (f32, BTHWC)
// EPI 3: MLP1 + exact GELU -> f16 row-major (M x HID)
// EPI 4: MLP2 + residual -> f32 row-major (M x C)
template <int KDIM, int EPI, int NOUT>
__global__ void gemm_k(const _Float16* __restrict__ A, const _Float16* __restrict__ Bp,
                       const float* __restrict__ bias, const float* __restrict__ resid,
                       _Float16* __restrict__ out16, float* __restrict__ out32) {
  constexpr int KT = KDIM / 32;
  constexpr int NT32 = NOUT / 32;
  const int lane = threadIdx.x & 31;
  const int flat = blockIdx.x * 8 + (threadIdx.x >> 5);
  const int mt32 = flat / NT32;
  const int nt32 = flat % NT32;
  if (mt32 >= C_ROWS / 32) return;
  const int mbase = mt32 * 32, nbase = nt32 * 32;

  v8f acc[2][2] = {};
  const _Float16* A0 = A + (size_t)mbase * KDIM;
  const _Float16* A1 = A + (size_t)(mbase + 16) * KDIM;
  for (int kt = 0; kt < KT; ++kt) {
    v16h a0 = load_a_frag(A0 + kt * 32, KDIM, lane);
    v16h a1 = load_a_frag(A1 + kt * 32, KDIM, lane);
    v16h b0 = *(const v16h*)(Bp + ((size_t)((nt32 * 2 + 0) * KT + kt) * 32 + lane) * 16);
    v16h b1 = *(const v16h*)(Bp + ((size_t)((nt32 * 2 + 1) * KT + kt) * 32 + lane) * 16);
    acc[0][0] = wmma16(a0, b0, acc[0][0]);
    acc[0][1] = wmma16(a0, b1, acc[0][1]);
    acc[1][0] = wmma16(a1, b0, acc[1][0]);
    acc[1][1] = wmma16(a1, b1, acc[1][1]);
  }

  const int cc = lane & 15;
  const int rb = (lane >> 4) * 8;
#pragma unroll
  for (int mt = 0; mt < 2; ++mt) {
#pragma unroll
    for (int g = 0; g < 8; ++g) {
      const int mrow = mbase + mt * 16 + rb + g;
#pragma unroll
      for (int nt = 0; nt < 2; ++nt) {
        const int col = nbase + nt * 16 + cc;
        float f = acc[mt][nt][g] + bias[col];
        if constexpr (EPI == 0) {
          int bwt = mrow / C_N, n = mrow - bwt * C_N;
          int head = col >> 5, hd = col & 31;
          out16[(((size_t)bwt * C_HEADS + head) * 32 + n) * 32 + hd] = (_Float16)f;
        } else if constexpr (EPI == 1) {
          int bwt = mrow / C_N, n = mrow - bwt * C_N;
          int head = col >> 5, hd = col & 31;
          out16[(((size_t)bwt * C_HEADS + head) * 32 + hd) * 32 + n] = (_Float16)f;
        } else if constexpr (EPI == 2) {
          int bwt = mrow / C_N, n = mrow - bwt * C_N;
          int b = bwt / (C_NW * C_T), r2 = bwt - b * (C_NW * C_T);
          int nwi = r2 >> 3, t = r2 & 7;
          int whi = nwi / 10, wwi = nwi - whi * 10;
          int i = n / 6, j = n - i * 6;
          int h = whi * C_WH + i + C_SH; if (h >= C_H) h -= C_H;
          int w = wwi * C_WW + j + C_SW; if (w >= C_Wd) w -= C_Wd;
          size_t xi = ((((size_t)b * C_T + t) * C_H + h) * C_Wd + w) * C_C + col;
          out32[xi] = resid[xi] + f;
        } else if constexpr (EPI == 3) {
          float y = 0.5f * f * (1.0f + erff(f * 0.70710678118654752f));
          out16[(size_t)mrow * C_HID + col] = (_Float16)y;
        } else {
          size_t idx = (size_t)mrow * C_C + col;
          out32[idx] = resid[idx] + f;
        }
      }
    }
  }
}

// ---------------- attention: one block = one (window,time), one wave = one head ----------------
__global__ void attn_k(const _Float16* __restrict__ q16, const _Float16* __restrict__ k16,
                       const _Float16* __restrict__ vT16, const float* __restrict__ rel_bias,
                       _Float16* __restrict__ out16) {
  __shared__ float s_sc[C_HEADS][32][32];
  __shared__ __align__(32) _Float16 s_at[C_HEADS][32][32];
  __shared__ float s_rb[99 * C_HEADS];
  const int bwt = blockIdx.x;
  const int head = threadIdx.x >> 5;
  const int lane = threadIdx.x & 31;
  for (int u = threadIdx.x; u < 99 * C_HEADS; u += 256) s_rb[u] = rel_bias[u];

  const size_t base = ((size_t)bwt * C_HEADS + head) * 1024;  // 32 tok x 32 hd
  const _Float16* qb = q16 + base;
  const _Float16* kb = k16 + base;
  const _Float16* vb = vT16 + base;

  // scores = q @ k^T   (K = HD = 32, one WMMA per 16x16 tile)
  v16h a0 = load_a_frag(qb, 32, lane);
  v16h a1 = load_a_frag(qb + 16 * 32, 32, lane);
  v16h b0 = load_b_kcont(kb, 32, lane);
  v16h b1 = load_b_kcont(kb + 16 * 32, 32, lane);
  v8f s00 = {}, s01 = {}, s10 = {}, s11 = {};
  s00 = wmma16(a0, b0, s00); s01 = wmma16(a0, b1, s01);
  s10 = wmma16(a1, b0, s10); s11 = wmma16(a1, b1, s11);
  const int cc = lane & 15, rb_ = (lane >> 4) * 8;
#pragma unroll
  for (int g = 0; g < 8; ++g) {
    s_sc[head][rb_ + g][cc]        = s00[g];
    s_sc[head][rb_ + g][cc + 16]   = s01[g];
    s_sc[head][16 + rb_ + g][cc]      = s10[g];
    s_sc[head][16 + rb_ + g][cc + 16] = s11[g];
  }
  __syncthreads();

  // softmax with scale + relative-position bias + shift mask; one row per lane
  {
    const int row = lane;
    int nwi = (bwt >> 3) % C_NW;
    int whi = nwi / 10, wwi = nwi - whi * 10;
    if (row < C_N) {
      int i1 = row / 6, j1 = row - i1 * 6;
      int hs1 = whi * C_WH + i1, ws1 = wwi * C_WW + j1;
      int reg1 = (hs1 < 55 ? 0 : (hs1 < 58 ? 1 : 2)) * 3 +
                 (ws1 < 54 ? 0 : (ws1 < 57 ? 1 : 2));
      float vals[C_N];
      float mx = -1e30f;
#pragma unroll
      for (int col = 0; col < C_N; ++col) {
        int i2 = col / 6, j2 = col - i2 * 6;
        int hs2 = whi * C_WH + i2, ws2 = wwi * C_WW + j2;
        int reg2 = (hs2 < 55 ? 0 : (hs2 < 58 ? 1 : 2)) * 3 +
                   (ws2 < 54 ? 0 : (ws2 < 57 ? 1 : 2));
        int ridx = (i1 - i2 + C_WH - 1) * (2 * C_WW - 1) + (j1 - j2 + C_WW - 1);
        float sc = s_sc[head][row][col] * C_SCALE + s_rb[ridx * C_HEADS + head] +
                   (reg1 == reg2 ? 0.f : -100.f);
        vals[col] = sc;
        mx = fmaxf(mx, sc);
      }
      float sum = 0.f;
#pragma unroll
      for (int col = 0; col < C_N; ++col) { vals[col] = __expf(vals[col] - mx); sum += vals[col]; }
      float inv = 1.0f / sum;
#pragma unroll
      for (int col = 0; col < C_N; ++col) s_at[head][row][col] = (_Float16)(vals[col] * inv);
      s_at[head][row][30] = (_Float16)0.f;
      s_at[head][row][31] = (_Float16)0.f;
    } else {
#pragma unroll
      for (int col = 0; col < 32; ++col) s_at[head][row][col] = (_Float16)0.f;
    }
  }
  __syncthreads();

  // out = attn @ v   (K = 32 tokens, v stored transposed so B is K-contiguous)
  v16h p0 = load_a_frag(&s_at[head][0][0], 32, lane);
  v16h p1 = load_a_frag(&s_at[head][16][0], 32, lane);
  v16h w0 = load_b_kcont(vb, 32, lane);
  v16h w1 = load_b_kcont(vb + 16 * 32, 32, lane);
  v8f o00 = {}, o01 = {}, o10 = {}, o11 = {};
  o00 = wmma16(p0, w0, o00); o01 = wmma16(p0, w1, o01);
  o10 = wmma16(p1, w0, o10); o11 = wmma16(p1, w1, o11);
#pragma unroll
  for (int g = 0; g < 8; ++g) {
#pragma unroll
    for (int mt = 0; mt < 2; ++mt) {
      int row = mt * 16 + rb_ + g;
      if (row < C_N) {
        size_t ob = ((size_t)bwt * C_N + row) * C_C + head * C_HD;
        out16[ob + cc]      = (_Float16)(mt == 0 ? o00[g] : o10[g]);
        out16[ob + 16 + cc] = (_Float16)(mt == 0 ? o01[g] : o11[g]);
      }
    }
  }
}

// ---------------- launch ----------------
extern "C" void kernel_launch(void* const* d_in, const int* in_sizes, int n_in,
                              void* d_out, int out_size, void* d_ws, size_t ws_size,
                              hipStream_t stream) {
  (void)in_sizes; (void)n_in; (void)out_size; (void)ws_size;
  const float* x   = (const float*)d_in[0];
  const float* ce  = (const float*)d_in[1];
  const float* g1  = (const float*)d_in[2];
  const float* be1 = (const float*)d_in[3];
  const float* Wq  = (const float*)d_in[4];
  const float* bq  = (const float*)d_in[5];
  const float* Wk  = (const float*)d_in[6];
  const float* bk  = (const float*)d_in[7];
  const float* Wv  = (const float*)d_in[8];
  const float* bv  = (const float*)d_in[9];
  const float* rb  = (const float*)d_in[10];
  const float* Wo  = (const float*)d_in[11];
  const float* bo  = (const float*)d_in[12];
  const float* g2  = (const float*)d_in[13];
  const float* be2 = (const float*)d_in[14];
  const float* W1  = (const float*)d_in[15];
  const float* b1  = (const float*)d_in[16];
  const float* W2  = (const float*)d_in[17];
  const float* b2  = (const float*)d_in[18];
  float* out = (float*)d_out;

  char* ws = (char*)d_ws;
  const size_t SZ_ACT = (size_t)C_ROWS * C_C * 2;                 // 29.5 MB
  const size_t SZ_QKV = (size_t)C_BWT * C_HEADS * 32 * 32 * 2;    // 31.5 MB
  const size_t SZ_X1  = (size_t)C_ROWS * C_C * 4;                 // 59 MB
  const size_t SZ_HID = (size_t)C_ROWS * C_HID * 2;               // 118 MB
  _Float16* wqk = (_Float16*)(ws);
  _Float16* w16 = (_Float16*)(ws + SZ_ACT);
  _Float16* q16 = (_Float16*)(ws + 2 * SZ_ACT);
  _Float16* k16 = (_Float16*)(ws + 2 * SZ_ACT + SZ_QKV);
  _Float16* v16 = (_Float16*)(ws + 2 * SZ_ACT + 2 * SZ_QKV);
  float*    x1  = (float*)   (ws + 2 * SZ_ACT + 3 * SZ_QKV);
  _Float16* hid = (_Float16*)(ws + 2 * SZ_ACT + 3 * SZ_QKV + SZ_X1);
  _Float16* wp  = (_Float16*)(ws + 2 * SZ_ACT + 3 * SZ_QKV + SZ_X1 + SZ_HID);
  _Float16* WqP = wp;
  _Float16* WkP = wp + 65536;
  _Float16* WvP = wp + 131072;
  _Float16* WoP = wp + 196608;
  _Float16* W1P = wp + 262144;
  _Float16* W2P = wp + 524288;
  _Float16* attn_o = wqk;  // reuse (wqk consumed by Q/K GEMMs)
  _Float16* x1n = w16;     // reuse (w16 consumed by V GEMM)

  pack_w<<<256, 256, 0, stream>>>(Wq, WqP, 8, 256, 65536);
  pack_w<<<256, 256, 0, stream>>>(Wk, WkP, 8, 256, 65536);
  pack_w<<<256, 256, 0, stream>>>(Wv, WvP, 8, 256, 65536);
  pack_w<<<256, 256, 0, stream>>>(Wo, WoP, 8, 256, 65536);
  pack_w<<<1024, 256, 0, stream>>>(W1, W1P, 8, 1024, 262144);
  pack_w<<<1024, 256, 0, stream>>>(W2, W2P, 32, 256, 262144);

  (void)hipMemsetAsync(v16, 0, SZ_QKV, stream);  // zero token-padding for V

  ln1_partition<<<C_ROWS, 256, 0, stream>>>(x, ce, g1, be1, wqk, w16);

  gemm_k<256, 0, 256><<<1800, 256, 0, stream>>>(wqk, WqP, bq, nullptr, q16, nullptr);
  gemm_k<256, 0, 256><<<1800, 256, 0, stream>>>(wqk, WkP, bk, nullptr, k16, nullptr);
  gemm_k<256, 1, 256><<<1800, 256, 0, stream>>>(w16, WvP, bv, nullptr, v16, nullptr);

  attn_k<<<C_BWT, 256, 0, stream>>>(q16, k16, v16, rb, attn_o);

  gemm_k<256, 2, 256><<<1800, 256, 0, stream>>>(attn_o, WoP, bo, x, nullptr, x1);

  ln2_k<<<C_ROWS, 256, 0, stream>>>(x1, g2, be2, x1n);

  gemm_k<256, 3, 1024><<<7200, 256, 0, stream>>>(x1n, W1P, b1, nullptr, hid, nullptr);
  gemm_k<1024, 4, 256><<<1800, 256, 0, stream>>>(hid, W2P, b2, x1, nullptr, out);
}
